// GNNReasoner_2534030704778
// MI455X (gfx1250) — matrix operand order repaired and checked by reading, not compiled, not run on hardware
//
#include <hip/hip_runtime.h>
#include <stdint.h>

typedef __attribute__((ext_vector_type(16))) _Float16 v16h;
typedef __attribute__((ext_vector_type(8)))  _Float16 v8h;
typedef __attribute__((ext_vector_type(8)))  float    v8f;
typedef __attribute__((ext_vector_type(4)))  float    v4f;

#define N_NODES 3072
#define F_IN    256
#define HID     128
#define HEADS   4
#define NC      5

static __device__ __forceinline__ v8f wmma_f16(v16h a, v16h b, v8f c) {
  // (neg_a, A, neg_b, B, c_mod, C, reuse_a, reuse_b)
  return __builtin_amdgcn_wmma_f32_16x16x32_f16(false, a, false, b, (short)0, c, false, false);
}

// CDNA5 async global->LDS copy (ASYNCcnt-tracked). VDST VGPR = LDS byte offset,
// VADDR = 64-bit global address (GV mode).
static __device__ __forceinline__ void async_load_b128_to_lds(uint32_t ldsOff, const void* gaddr) {
  asm volatile("global_load_async_to_lds_b128 %0, %1, off"
               :
               : "v"(ldsOff), "v"((uint64_t)(uintptr_t)gaddr)
               : "memory");
}
static __device__ __forceinline__ void wait_asynccnt0() {
  asm volatile("s_wait_asynccnt 0x0" ::: "memory");
}

// ---------------------------------------------------------------------------
// fp32 -> f16 conversion
__global__ void k_f32_to_f16(const float* __restrict__ src, _Float16* __restrict__ dst, int count) {
  int i = blockIdx.x * blockDim.x + threadIdx.x;
  if (i < count) dst[i] = (_Float16)src[i];
}

// W [H][Fin][Fh] fp32 -> WT16 [H][Fh][Fin] f16 (B-matrix friendly layout)
__global__ void k_w_transpose(const float* __restrict__ W, _Float16* __restrict__ WT,
                              int Fin, int Fh) {
  int i = blockIdx.x * blockDim.x + threadIdx.x;
  int total = HEADS * Fin * Fh;
  if (i >= total) return;
  int o = i % Fh;
  int f = (i / Fh) % Fin;
  int h = i / (Fh * Fin);
  WT[(h * Fh + o) * Fin + f] = (_Float16)W[i];
}

// ---------------------------------------------------------------------------
// h = x @ W per head via WMMA. One wave per (16-row block, head).
//   x16 : [N][FIN] f16        WT : [H][FH][FIN] f16
//   hT  : [H][FH][N] f16  (transposed output, B-matrix layout for aggregation)
template <int FIN, int FH>
__global__ void k_gemm_h(const _Float16* __restrict__ x16,
                         const _Float16* __restrict__ WT,
                         _Float16* __restrict__ hT) {
  constexpr int NT = FH / 16;               // # of 16-wide output tiles (1 or 2)
  const int lane = threadIdx.x & 31;
  const int wave = threadIdx.x >> 5;
  const int hlf  = lane >> 4;               // lane half (wave32: lanes 0-15 / 16-31)
  const int l16  = lane & 15;
  const int rb   = blockIdx.x * 4 + wave;   // row block index
  const int head = blockIdx.y;
  const int n0   = rb * 16;

  v8f acc[NT] = {};

  const _Float16* xrow  = x16 + (size_t)(n0 + l16) * FIN;
  const _Float16* wbase = WT + (size_t)head * FH * FIN;

#pragma unroll
  for (int k0 = 0; k0 < FIN; k0 += 32) {
    // A (16x32 f16): lane holds row M=l16; K = 16*hlf + e
    union { v16h v; v8h h2[2]; } A;
    A.h2[0] = *(const v8h*)(xrow + k0 + 16 * hlf);
    A.h2[1] = *(const v8h*)(xrow + k0 + 16 * hlf + 8);
#pragma unroll
    for (int ot = 0; ot < NT; ++ot) {
      // B (32x16 f16): lane holds col N=l16, K = 16*hlf + e (contiguous 32B)
      v16h B = *(const v16h*)(wbase + (size_t)(ot * 16 + l16) * FIN + k0 + 16 * hlf);
      acc[ot] = wmma_f16(A.v, B, acc[ot]);
    }
  }
  // D (16x16 f32): element r -> node n0 + r + 8*hlf, feature ot*16 + l16
#pragma unroll
  for (int ot = 0; ot < NT; ++ot) {
    _Float16* drow = hT + (size_t)(head * FH + ot * 16 + l16) * N_NODES + n0 + 8 * hlf;
#pragma unroll
    for (int r = 0; r < 8; ++r) drow[r] = (_Float16)acc[ot][r];
  }
}

// ---------------------------------------------------------------------------
// s1[h][n] = h[h][n][:] . a1[h],  s2 likewise (a packed as [H][2*FH][1])
template <int FH>
__global__ void k_scores(const _Float16* __restrict__ hT, const float* __restrict__ aVec,
                         float* __restrict__ s1, float* __restrict__ s2) {
  int i = blockIdx.x * blockDim.x + threadIdx.x;    // i = h*N + n
  if (i >= HEADS * N_NODES) return;
  int n = i % N_NODES;
  int h = i / N_NODES;
  float r1 = 0.f, r2 = 0.f;
#pragma unroll
  for (int o = 0; o < FH; ++o) {
    float hv = (float)hT[(size_t)(h * FH + o) * N_NODES + n];
    r1 += hv * aVec[h * 2 * FH + o];
    r2 += hv * aVec[h * 2 * FH + FH + o];
  }
  s1[i] = r1;
  s2[i] = r2;
}

// ---------------------------------------------------------------------------
// Masked-softmax stats: one wave per row, all 4 heads from ONE adj read.
// Online (max, sum) per lane, then wave32 shfl_xor tree combine.
__global__ void k_stats(const float* __restrict__ adj, const float* __restrict__ s1,
                        const float* __restrict__ s2, float* __restrict__ mOut,
                        float* __restrict__ invOut) {
  const int lane = threadIdx.x & 31;
  const int wave = threadIdx.x >> 5;
  const int row  = blockIdx.x * (blockDim.x >> 5) + wave;
  if (row >= N_NODES) return;

  float s1v[HEADS];
#pragma unroll
  for (int h = 0; h < HEADS; ++h) s1v[h] = s1[h * N_NODES + row];

  float m[HEADS], s[HEADS];
#pragma unroll
  for (int h = 0; h < HEADS; ++h) { m[h] = -1e30f; s[h] = 0.f; }

  const float* arow = adj + (size_t)row * N_NODES;
  for (int j = lane; j < N_NODES; j += 32) {
    float a = arow[j];
    if (a != 0.0f) {
#pragma unroll
      for (int h = 0; h < HEADS; ++h) {
        float e = s1v[h] + s2[h * N_NODES + j];
        e = (e > 0.f) ? e : 0.01f * e;               // leaky_relu(0.01)
        if (e > m[h]) { s[h] = s[h] * __expf(m[h] - e) + 1.0f; m[h] = e; }
        else          { s[h] += __expf(e - m[h]); }
      }
    }
  }
#pragma unroll
  for (int off = 16; off >= 1; off >>= 1) {
#pragma unroll
    for (int h = 0; h < HEADS; ++h) {
      float mo = __shfl_xor(m[h], off);
      float so = __shfl_xor(s[h], off);
      float mn = fmaxf(m[h], mo);
      float acc = 0.f;
      if (s[h] > 0.f) acc += s[h] * __expf(m[h] - mn);
      if (so   > 0.f) acc += so   * __expf(mo  - mn);
      m[h] = mn;
      s[h] = acc;
    }
  }
  if (lane == 0) {
#pragma unroll
    for (int h = 0; h < HEADS; ++h) {
      mOut[h * N_NODES + row]   = m[h];
      invOut[h * N_NODES + row] = (s[h] > 0.f) ? 1.0f / s[h] : 0.f;
    }
  }
}

// ---------------------------------------------------------------------------
// Flash-style aggregation: out[i] = sum_j alpha_ij * h_j, fused ELU + concat.
// Block = 128 threads = 4 waves = 4 heads sharing one 16-row block.
// adj tiles move via async global->LDS (double-buffered, load of tile t+1
// overlaps compute of tile t; one barrier per iteration).
template <int FH>
__global__ void k_aggregate(const float* __restrict__ adj,
                            const _Float16* __restrict__ hT,   // [H][FH][N]
                            const float* __restrict__ s1,
                            const float* __restrict__ s2,
                            const float* __restrict__ mIn,
                            const float* __restrict__ invIn,
                            _Float16* __restrict__ xNext16,    // [N][H*FH] f16
                            float* __restrict__ outF32) {      // optional [N][H*FH]
  constexpr int NT   = FH / 16;
  constexpr int OUTW = HEADS * FH;
  __shared__ __align__(16) float adjT[2][16 * 32];
  __shared__ __align__(16) float s2T[2][HEADS * 32];

  const int tid  = threadIdx.x;
  const int lane = tid & 31;
  const int head = tid >> 5;          // wave index == head
  const int hlf  = lane >> 4;
  const int l16  = lane & 15;
  const int i0   = blockIdx.x * 16;

  const int   myRow = i0 + l16;
  const float s1r = s1[head * N_NODES + myRow];
  const float mr  = mIn[head * N_NODES + myRow];
  const float ivr = invIn[head * N_NODES + myRow];

  v8f acc[NT] = {};

  // 128 threads stage the 16x32 fp32 adj tile as one b128 async copy each
  const int arow = tid >> 3;
  const int acol = (tid & 7) * 4;
  const uint32_t adjLdsOff[2] = {
    (uint32_t)(uintptr_t)&adjT[0][arow * 32 + acol],
    (uint32_t)(uintptr_t)&adjT[1][arow * 32 + acol],
  };
  const float* adjSrc = adj + (size_t)(i0 + arow) * N_NODES + acol;
  const _Float16* hb  = hT + (size_t)head * FH * N_NODES;

  // prologue: stage tile 0
  async_load_b128_to_lds(adjLdsOff[0], adjSrc);
  s2T[0][head * 32 + lane] = s2[head * N_NODES + lane];

  for (int j0 = 0, it = 0; j0 < N_NODES; j0 += 32, ++it) {
    const int buf = it & 1;
    wait_asynccnt0();                 // my async write into adjT[buf] is done
    __syncthreads();                  // everyone's writes visible

    if (j0 + 32 < N_NODES) {          // stage tile t+1 (overlaps compute below)
      async_load_b128_to_lds(adjLdsOff[buf ^ 1], adjSrc + j0 + 32);
      s2T[buf ^ 1][head * 32 + lane] = s2[head * N_NODES + j0 + 32 + lane];
    }

    // build probability tile P directly in WMMA A layout
    union { v16h v; _Float16 e[16]; } P;
#pragma unroll
    for (int e = 0; e < 16; ++e) {
      int jl = e + 16 * hlf;
      float a = adjT[buf][l16 * 32 + jl];
      float p = 0.f;
      if (a != 0.f) {
        float ev = s1r + s2T[buf][head * 32 + jl];
        ev = (ev > 0.f) ? ev : 0.01f * ev;          // leaky_relu(0.01)
        p = __expf(ev - mr) * ivr;
      }
      P.e[e] = (_Float16)p;
    }
#pragma unroll
    for (int ot = 0; ot < NT; ++ot) {
      v16h B = *(const v16h*)(hb + (size_t)(ot * 16 + l16) * N_NODES + j0 + 16 * hlf);
      acc[ot] = wmma_f16(P.v, B, acc[ot]);
    }
  }

  // epilogue: ELU, concat heads, write f16 (next layer) and optional f32
#pragma unroll
  for (int ot = 0; ot < NT; ++ot) {
#pragma unroll
    for (int r = 0; r < 8; ++r) {
      int n = i0 + r + 8 * hlf;
      int o = head * FH + ot * 16 + l16;
      float v = acc[ot][r];
      float elu = (v > 0.f) ? v : (__expf(v) - 1.0f);
      xNext16[(size_t)n * OUTW + o] = (_Float16)elu;
      if (outF32) outF32[(size_t)n * OUTW + o] = elu;
    }
  }
}

// ---------------------------------------------------------------------------
// Final small MLP heads (node classifier + anomaly): one thread per node.
__global__ void k_mlp(const float* __restrict__ h3,
                      const float* __restrict__ nc_w1, const float* __restrict__ nc_b1,
                      const float* __restrict__ nc_w2, const float* __restrict__ nc_b2,
                      const float* __restrict__ ad_w1, const float* __restrict__ ad_b1,
                      const float* __restrict__ ad_w2, const float* __restrict__ ad_b2,
                      float* __restrict__ logits, float* __restrict__ anomaly) {
  int n = blockIdx.x * blockDim.x + threadIdx.x;
  if (n >= N_NODES) return;
  const float* hr = h3 + (size_t)n * 64;

  float t1[32];
  for (int k = 0; k < 32; ++k) {
    float acc = nc_b1[k];
    for (int f = 0; f < 64; ++f) acc += hr[f] * nc_w1[f * 32 + k];
    t1[k] = fmaxf(acc, 0.f);
  }
  for (int c = 0; c < NC; ++c) {
    float acc = nc_b2[c];
    for (int k = 0; k < 32; ++k) acc += t1[k] * nc_w2[k * NC + c];
    logits[n * NC + c] = acc;
  }
  float t2[32];
  for (int k = 0; k < 32; ++k) {
    float acc = ad_b1[k];
    for (int f = 0; f < 64; ++f) acc += hr[f] * ad_w1[f * 32 + k];
    t2[k] = fmaxf(acc, 0.f);
  }
  float acc = ad_b2[0];
  for (int k = 0; k < 32; ++k) acc += t2[k] * ad_w2[k];
  anomaly[n] = 1.f / (1.f + __expf(-acc));
}

// ---------------------------------------------------------------------------
template <int FIN, int FH>
static void run_layer(const float* W, const float* a, const _Float16* xin,
                      _Float16* xout, float* f32out,
                      const float* adj, _Float16* WT16, _Float16* hT16,
                      float* s1b, float* s2b, float* mb, float* invb,
                      hipStream_t stream) {
  int wtot = HEADS * FIN * FH;
  k_w_transpose<<<(wtot + 255) / 256, 256, 0, stream>>>(W, WT16, FIN, FH);
  dim3 gg(N_NODES / 16 / 4, HEADS);
  k_gemm_h<FIN, FH><<<gg, 128, 0, stream>>>(xin, WT16, hT16);
  k_scores<FH><<<(HEADS * N_NODES + 255) / 256, 256, 0, stream>>>(hT16, a, s1b, s2b);
  k_stats<<<N_NODES / 4, 128, 0, stream>>>(adj, s1b, s2b, mb, invb);
  k_aggregate<FH><<<N_NODES / 16, 128, 0, stream>>>(adj, hT16, s1b, s2b, mb, invb,
                                                    xout, f32out);
}

extern "C" void kernel_launch(void* const* d_in, const int* in_sizes, int n_in,
                              void* d_out, int out_size, void* d_ws, size_t ws_size,
                              hipStream_t stream) {
  (void)in_sizes; (void)n_in; (void)out_size; (void)ws_size;

  const float* x     = (const float*)d_in[0];
  const float* adj   = (const float*)d_in[1];
  const float* W1    = (const float*)d_in[2];
  const float* a1    = (const float*)d_in[3];
  const float* W2    = (const float*)d_in[4];
  const float* a2    = (const float*)d_in[5];
  const float* W3    = (const float*)d_in[6];
  const float* a3    = (const float*)d_in[7];
  const float* nc_w1 = (const float*)d_in[8];
  const float* nc_b1 = (const float*)d_in[9];
  const float* nc_w2 = (const float*)d_in[10];
  const float* nc_b2 = (const float*)d_in[11];
  const float* ad_w1 = (const float*)d_in[12];
  const float* ad_b1 = (const float*)d_in[13];
  const float* ad_w2 = (const float*)d_in[14];
  const float* ad_b2 = (const float*)d_in[15];

  char* ws = (char*)d_ws;
  size_t off = 0;
  auto alloc = [&](size_t bytes) -> void* {
    void* p = ws + off;
    off = (off + bytes + 255) & ~(size_t)255;
    return p;
  };
  _Float16* x16a = (_Float16*)alloc((size_t)N_NODES * F_IN * 2);
  _Float16* x16b = (_Float16*)alloc((size_t)N_NODES * HID * 2);
  _Float16* WT16 = (_Float16*)alloc((size_t)HEADS * 32 * 256 * 2);
  _Float16* hT16 = (_Float16*)alloc((size_t)HEADS * 32 * N_NODES * 2);
  float* s1b  = (float*)alloc((size_t)HEADS * N_NODES * 4);
  float* s2b  = (float*)alloc((size_t)HEADS * N_NODES * 4);
  float* mb   = (float*)alloc((size_t)HEADS * N_NODES * 4);
  float* invb = (float*)alloc((size_t)HEADS * N_NODES * 4);

  float* outH3 = (float*)d_out;                           // [N][64]
  float* outLg = outH3 + (size_t)N_NODES * 64;            // [N][5]
  float* outAn = outLg + (size_t)N_NODES * NC;            // [N]

  k_f32_to_f16<<<(N_NODES * F_IN + 255) / 256, 256, 0, stream>>>(x, x16a, N_NODES * F_IN);

  run_layer<F_IN, 32>(W1, a1, x16a, x16b, nullptr, adj, WT16, hT16, s1b, s2b, mb, invb, stream);
  run_layer<HID,  32>(W2, a2, x16b, x16a, nullptr, adj, WT16, hT16, s1b, s2b, mb, invb, stream);
  run_layer<HID,  16>(W3, a3, x16a, x16b, outH3,  adj, WT16, hT16, s1b, s2b, mb, invb, stream);

  k_mlp<<<(N_NODES + 255) / 256, 256, 0, stream>>>(outH3, nc_w1, nc_b1, nc_w2, nc_b2,
                                                   ad_w1, ad_b1, ad_w2, ad_b2,
                                                   outLg, outAn);
}
